// Model_39676907883434
// MI455X (gfx1250) — matrix operand order repaired and checked by reading, not compiled, not run on hardware
//
#include <hip/hip_runtime.h>
#include <hip/hip_bf16.h>

typedef __bf16 v16bf __attribute__((ext_vector_type(16)));
typedef float  v8f   __attribute__((ext_vector_type(8)));
typedef unsigned v4u __attribute__((ext_vector_type(4)));
typedef int      v8i __attribute__((ext_vector_type(8)));
typedef int      v4i __attribute__((ext_vector_type(4)));

#define BATCH 32
#define SEQ   2048
#define DIM   128
#define QB    64     // query rows per block
#define KC    64     // kv rows per chunk
#define NW    4      // waves per block (wave32)
#define SZ    (BATCH * SEQ * DIM)   // 8388608 elements per tensor

#define HAVE_TDM __has_builtin(__builtin_amdgcn_tensor_load_to_lds)

__device__ __forceinline__ unsigned short f2bf(float f) {
    unsigned u = __builtin_bit_cast(unsigned, f);
    u += 0x7FFFu + ((u >> 16) & 1u);   // round-to-nearest-even
    return (unsigned short)(u >> 16);
}

// (b,q,k) packs exactly into 32 bits: b<<22 | q<<11 | k   (S = 2048 = 2^11)
__device__ __forceinline__ unsigned drophash(unsigned idx) {
    unsigned h = idx * 0x9E3779B9u;
    h ^= h >> 16;
    h *= 0x85EBCA6Bu;
    h ^= h >> 13;
    return h;
}

union FragBF { unsigned u[8]; v16bf v; };

__device__ __forceinline__ unsigned lds_offset(const void* p) {
    // generic -> LDS address space -> 32-bit LDS byte offset (for TDM D#)
    return (unsigned)(unsigned long long)(const __attribute__((address_space(3))) char*)p;
}

#if HAVE_TDM
// Issue one 2-D bf16 tile load global->LDS via the Tensor Data Mover.
//   tile: tile_d0 x tile_d1 elements, row stride in the tensor = stride0 elements
__device__ __forceinline__ void tdm_load_2d(unsigned lds_addr, const unsigned short* gptr,
                                            unsigned tensor_d0, unsigned tensor_d1,
                                            unsigned tile_d0, unsigned tile_d1,
                                            unsigned stride0) {
    unsigned long long ga = (unsigned long long)gptr;   // byte address, fits 57 bits
    v4u g0;
    g0[0] = 1u;                                         // count=1, is_restore=0, gather=0
    g0[1] = lds_addr;                                   // lds_addr[31:0]
    g0[2] = (unsigned)(ga & 0xFFFFFFFFu);               // global_addr[31:0]
    g0[3] = (unsigned)((ga >> 32) & 0x01FFFFFFu) | 0x80000000u;  // addr[56:32] | type=2

    v8i g1;
    g1[0] = (int)0x00010000u;                           // data_size=1 (2 bytes), mask=0
    g1[1] = (int)((tensor_d0 & 0xFFFFu) << 16);         // [63:48] tensor_dim0[15:0]
    g1[2] = (int)(((tensor_d0 >> 16) & 0xFFFFu) | ((tensor_d1 & 0xFFFFu) << 16));
    g1[3] = (int)(((tensor_d1 >> 16) & 0xFFFFu) | (tile_d0 << 16));   // [127:112] tile_dim0
    g1[4] = (int)(tile_d1 & 0xFFFFu);                   // [143:128] tile_dim1, tile_dim2=0
    g1[5] = (int)stride0;                               // tensor_dim0_stride[31:0]
    g1[6] = 0;                                          // stride0[47:32]=0, stride1[15:0]=0
    g1[7] = 0;
    v4i zero4 = {0, 0, 0, 0};
    v8i zero8 = {0, 0, 0, 0, 0, 0, 0, 0};
    // clang-23 / therock-10.0 arity: (g0 v4u, g1 v8i, g2 v4i, g3 v4i, v8i, i32 cpol)
    __builtin_amdgcn_tensor_load_to_lds(g0, g1, zero4, zero4, zero8, 0);
}
#endif

// ---- pre-pass: f32 -> bf16 (Q scaled, K row-major, V transposed per batch) ----
__global__ __launch_bounds__(256)
void prep_kernel(const float* __restrict__ x1, const float* __restrict__ x2,
                 unsigned short* __restrict__ qbf, unsigned short* __restrict__ kbf,
                 unsigned short* __restrict__ vtbf) {
    const float scale = 3.3635856610148585f;            // 128^0.25
    size_t i = (size_t)blockIdx.x * 256 + threadIdx.x;  // one element each, SZ total
    unsigned b   = (unsigned)(i >> 18);                 // SEQ*DIM = 2^18
    unsigned rem = (unsigned)(i & 0x3FFFFu);
    unsigned row = rem >> 7;                            // 0..2047
    unsigned d   = rem & 127u;
    qbf[i] = f2bf(x1[i] * scale);
    float kv = x2[i];
    kbf[i] = f2bf(kv);
    vtbf[((size_t)b << 18) + ((size_t)d << 11) + row] = f2bf(kv);   // Vt[b][d][k], L2 absorbs scatter
}

__global__ __launch_bounds__(128)
void fattn_bf16_wmma(const unsigned short* __restrict__ qbf,
                     const unsigned short* __restrict__ kbf,
                     const unsigned short* __restrict__ vtbf,
                     float* __restrict__ out) {
    __shared__ unsigned short sK[2][KC * DIM];    // 2 x 16 KB, bf16 row-major (double buffer)
    __shared__ unsigned short sVt[2][DIM * KC];   // 2 x 16 KB, bf16 transposed [d][k]
    __shared__ unsigned short sP[NW][16 * KC];    //     8 KB, per-wave P staging

    const int tid  = threadIdx.x;
    const int wave = tid >> 5;
    const int lane = tid & 31;
    const int lo   = lane & 15;
    const int hi   = lane >> 4;

    const int b     = blockIdx.x >> 5;            // 32 q-blocks per batch
    const int qblk  = blockIdx.x & 31;
    const int qbase = qblk * QB;

    const float    inv_keep = 1.0f / 0.9f;
    const unsigned thresh   = 429496729u;         // 0.1 * 2^32

    const unsigned short* Qg = qbf + ((size_t)b * SEQ + qbase) * DIM;
    const unsigned short* Kg = kbf + ((size_t)b * SEQ) * DIM;
    const unsigned short* Vg = vtbf + ((size_t)b * DIM) * SEQ;

    // Issue the chunk load for [kv0, kv0+KC) into buffer `buf`.
    auto issue_load = [&](int kv0, int buf) {
#if HAVE_TDM
        // 4 waves issue 4 TDM descriptors: K halves + Vt halves
        if (wave < 2) {
            tdm_load_2d(lds_offset(&sK[buf][0]) + (unsigned)(wave * 32 * DIM * 2),
                        Kg + (size_t)(kv0 + wave * 32) * DIM,
                        DIM, SEQ, DIM, 32, DIM);
        } else {
            int d0 = (wave - 2) * 64;
            tdm_load_2d(lds_offset(&sVt[buf][0]) + (unsigned)(d0 * KC * 2),
                        Vg + (size_t)d0 * SEQ + kv0,
                        SEQ, DIM, KC, 64, SEQ);
        }
#else
        #pragma unroll
        for (int i = 0; i < (KC * DIM) / (128 * 8); ++i) {   // 8 iters
            int v   = tid + i * 128;
            int row = v >> 4;                                // 16 x uint4 per 128-elem row
            int col = (v & 15) * 8;
            uint4 kd = *(const uint4*)&Kg[(size_t)(kv0 + row) * DIM + col];
            *(uint4*)&sK[buf][row * DIM + col] = kd;
        }
        #pragma unroll
        for (int i = 0; i < (DIM * KC) / (128 * 8); ++i) {   // 8 iters
            int v   = tid + i * 128;
            int d   = v >> 3;                                // 8 x uint4 per 64-elem row
            int col = (v & 7) * 8;
            uint4 vd = *(const uint4*)&Vg[(size_t)d * SEQ + kv0 + col];
            *(uint4*)&sVt[buf][d * KC + col] = vd;
        }
#endif
    };

    // ---- Q A-fragments straight from global (pre-scaled bf16), loop-invariant ----
    FragBF qf[4];
    #pragma unroll
    for (int ks = 0; ks < 4; ++ks) {
        #pragma unroll
        for (int j = 0; j < 8; ++j) {
            int k = 32 * ks + ((j < 4) ? (hi * 8 + 2 * j) : (16 + hi * 8 + 2 * (j - 4)));
            qf[ks].u[j] = *(const unsigned*)&Qg[(size_t)(wave * 16 + lo) * DIM + k];
        }
    }

    v8f acc[8];
    #pragma unroll
    for (int g = 0; g < 8; ++g)
        #pragma unroll
        for (int r = 0; r < 8; ++r) acc[g][r] = 0.0f;

    float m_i[8], l_i[8];                         // l_i: per-lane PARTIAL row sums
    #pragma unroll
    for (int r = 0; r < 8; ++r) { m_i[r] = -3.0e38f; l_i[r] = 0.0f; }

    const unsigned idx_b = ((unsigned)b << 22) | (unsigned)lo;

    issue_load(0, 0);                             // prologue: prime buffer 0

    for (int chunk = 0; chunk < SEQ / KC; ++chunk) {
        const int kv0 = chunk * KC;
        const int cur = chunk & 1;

#if HAVE_TDM
        __builtin_amdgcn_s_wait_tensorcnt(0);     // own DMA for `cur` complete
#endif
        __syncthreads();                          // all waves' DMAs/stores visible;
                                                  // all reads of buffer cur^1 finished
        if (kv0 + KC < SEQ)
            issue_load(kv0 + KC, cur ^ 1);        // overlap next DMA with this compute

        // ---- scores: S[16x64] = Qtile(16x128) x K_chunk^T, 4 n-tiles ----
        v8f sc[4];
        #pragma unroll
        for (int t = 0; t < 4; ++t) {
            v8f c = {};
            #pragma unroll
            for (int ks = 0; ks < 4; ++ks) {
                FragBF kb;
                #pragma unroll
                for (int j = 0; j < 8; ++j) {
                    int k = 32 * ks + ((j < 4) ? (hi * 8 + 2 * j) : (16 + hi * 8 + 2 * (j - 4)));
                    kb.u[j] = *(const unsigned*)&sK[cur][(16 * t + lo) * DIM + k];
                }
                c = __builtin_amdgcn_wmma_f32_16x16x32_bf16(
                        false, qf[ks].v, false, kb.v, (short)0, c, false, false);
            }
            sc[t] = c;
        }

        // ---- online softmax (row-max reduce only) + dropout, stage P ----
        float alpha[8];
        #pragma unroll
        for (int r = 0; r < 8; ++r) {
            float rm = fmaxf(fmaxf(sc[0][r], sc[1][r]), fmaxf(sc[2][r], sc[3][r]));
            #pragma unroll
            for (int m = 1; m < 16; m <<= 1) rm = fmaxf(rm, __shfl_xor(rm, m, 32));
            float mnew = fmaxf(m_i[r], rm);
            alpha[r] = __expf(m_i[r] - mnew);
            m_i[r] = mnew;

            unsigned qrow = (unsigned)(qbase + wave * 16 + r + 8 * hi);
            unsigned ib = idx_b | (qrow << 11) | (unsigned)kv0;

            float psum = 0.0f;
            #pragma unroll
            for (int t = 0; t < 4; ++t) {
                float p = __expf(sc[t][r] - mnew);
                psum += p;                                     // pre-dropout (softmax denom)
                p = (drophash(ib + 16u * t) > thresh) ? p : 0.0f;
                sP[wave][(r + 8 * hi) * KC + 16 * t + lo] = f2bf(p);
            }
            l_i[r] = l_i[r] * alpha[r] + psum;                 // per-lane partial
        }

        // ---- rescale running accumulator ----
        #pragma unroll
        for (int g = 0; g < 8; ++g)
            #pragma unroll
            for (int r = 0; r < 8; ++r) acc[g][r] *= alpha[r];

        // ---- P A-fragments (two 16x32 k-slices of the 16x64 P tile) ----
        FragBF pf[2];
        #pragma unroll
        for (int s = 0; s < 2; ++s)
            #pragma unroll
            for (int j = 0; j < 8; ++j) {
                int k = 32 * s + ((j < 4) ? (hi * 8 + 2 * j) : (16 + hi * 8 + 2 * (j - 4)));
                pf[s].u[j] = *(const unsigned*)&sP[wave][lo * KC + k];
            }

        // ---- O += P(16x64) x V(64x128), 8 d-groups x 2 k-slices ----
        #pragma unroll
        for (int g = 0; g < 8; ++g) {
            #pragma unroll
            for (int s = 0; s < 2; ++s) {
                FragBF vb;
                #pragma unroll
                for (int j = 0; j < 8; ++j) {
                    int k = 32 * s + ((j < 4) ? (hi * 8 + 2 * j) : (16 + hi * 8 + 2 * (j - 4)));
                    vb.u[j] = *(const unsigned*)&sVt[cur][(16 * g + lo) * KC + k];
                }
                acc[g] = __builtin_amdgcn_wmma_f32_16x16x32_bf16(
                             false, pf[s].v, false, vb.v, (short)0, acc[g], false, false);
            }
        }
    }

    // ---- epilogue: reduce per-lane partial denominators ONCE, normalize, store ----
    float inv[8];
    #pragma unroll
    for (int r = 0; r < 8; ++r) {
        float ls = l_i[r];
        #pragma unroll
        for (int m = 1; m < 16; m <<= 1) ls += __shfl_xor(ls, m, 32);
        inv[r] = inv_keep / ls;                    // dropout 1/(1-p) folded in here
    }

    float* Og = out + ((size_t)b * SEQ + qbase + wave * 16) * DIM;
    #pragma unroll
    for (int g = 0; g < 8; ++g)
        #pragma unroll
        for (int r = 0; r < 8; ++r)
            Og[(size_t)(r + 8 * hi) * DIM + 16 * g + lo] = acc[g][r] * inv[r];
}

extern "C" void kernel_launch(void* const* d_in, const int* in_sizes, int n_in,
                              void* d_out, int out_size, void* d_ws, size_t ws_size,
                              hipStream_t stream) {
    (void)in_sizes; (void)n_in; (void)out_size; (void)ws_size;
    const float* x1 = (const float*)d_in[0];
    const float* x2 = (const float*)d_in[1];
    float* out = (float*)d_out;

    unsigned short* qbf  = (unsigned short*)d_ws;                 // 16 MB
    unsigned short* kbf  = qbf + SZ;                              // 16 MB
    unsigned short* vtbf = kbf + SZ;                              // 16 MB  (48 MB total)

    prep_kernel<<<SZ / 256, 256, 0, stream>>>(x1, x2, qbf, kbf, vtbf);

    dim3 grid(BATCH * (SEQ / QB));   // 32 batches x 32 q-blocks = 1024
    dim3 block(128);                 // 4 wave32 waves
    fattn_bf16_wmma<<<grid, block, 0, stream>>>(qbf, kbf, vtbf, out);
}
